// GraphNetwork_76836964925800
// MI455X (gfx1250) — matrix operand order repaired and checked by reading, not compiled, not run on hardware
//
#include <hip/hip_runtime.h>
#include <hip/hip_bf16.h>
#include <math.h>

// ---------------------------------------------------------------------------
// Static problem dims (match reference)
// ---------------------------------------------------------------------------
#define NNODES 4096
#define FIN    512
#define HID    256
#define NREL   16
#define NEDGE  65536
#define BB     32
#define SS     128
#define DH     768           // FIN + HID
#define D2     1536          // 2*DH
#define G4     3072          // 4*DH

typedef __attribute__((ext_vector_type(16))) _Float16 v16h;
typedef __attribute__((ext_vector_type(8)))  float    v8f;

#define FLAG_BIAS   1
#define FLAG_RELU   2
#define FLAG_OUTF16 4
#define FLAG_OUTF32 8
#define FLAG_ACCUM  16

#if defined(__HIP_DEVICE_COMPILE__) && __has_builtin(__builtin_amdgcn_tensor_load_to_lds)
#define USE_TDM 1
#else
#define USE_TDM 0
#endif

__device__ __forceinline__ float sigmoidf_(float x) { return 1.0f / (1.0f + __expf(-x)); }

#define WMMA16(a, b, c) \
  __builtin_amdgcn_wmma_f32_16x16x32_f16(false, (a), false, (b), (short)0, (c), false, false)

// ---------------------------------------------------------------------------
// Tensor Data Mover: issue a 2D tile load (rows x 64 halves, row stride
// strideElems) into LDS with hardware padding of 4 dwords every 32 dwords,
// i.e. a padded 72-half LDS row stride. Descriptor per CDNA5 ISA ch.8.
// ---------------------------------------------------------------------------
#if USE_TDM
typedef __attribute__((ext_vector_type(4))) unsigned int u32x4;
typedef __attribute__((ext_vector_type(8))) int i32x8;
typedef __attribute__((ext_vector_type(4))) int i32x4;

__device__ __forceinline__ void tdm_load_2d(const _Float16* gsrc, unsigned lds_off,
                                            long strideElems, int rows) {
  unsigned long long ga = (unsigned long long)(uintptr_t)gsrc;
  u32x4 g0;
  g0[0] = 1u;                                             // count=1, user mode
  g0[1] = lds_off;                                        // lds_addr (bytes)
  g0[2] = (unsigned)ga;                                   // global_addr[31:0]
  g0[3] = (unsigned)((ga >> 32) & 0x01ffffffull)          // global_addr[56:32]
          | (2u << 30);                                   // type = 2 ("image")
  i32x8 g1;
  g1[0] = (1 << 16)                                       // data_size = 2 bytes
        | (1 << 20)                                       // pad_enable
        | (4 << 22)                                       // pad_interval: 32 dwords
        | (3 << 25);                                      // pad_amount: 4 dwords
  g1[1] = (int)(64u << 16);                               // tensor_dim0 = 64 (lo16)
  g1[2] = (int)((unsigned)rows << 16);                    // dim0 hi | tensor_dim1 lo16
  g1[3] = (int)(64u << 16);                               // dim1 hi | tile_dim0 = 64
  g1[4] = rows & 0xffff;                                  // tile_dim1 | tile_dim2 = 0
  g1[5] = (int)(unsigned)(strideElems & 0xffffffffl);     // tensor_dim0_stride lo32
  g1[6] = (int)(unsigned)((strideElems >> 32) & 0xffffl); // stride hi16 | dim1_stride=0
  g1[7] = 0;
  i32x4 gz = {0, 0, 0, 0};                                // groups 2/3: dims 3/4 unused
#if __clang_major__ >= 23
  i32x8 gz8 = {0, 0, 0, 0, 0, 0, 0, 0};
  __builtin_amdgcn_tensor_load_to_lds(g0, g1, gz, gz, gz8, 0);
#else
  __builtin_amdgcn_tensor_load_to_lds(g0, g1, gz, gz, 0);
#endif
}
#endif

// ---------------------------------------------------------------------------
// Elementwise helpers
// ---------------------------------------------------------------------------
__global__ __launch_bounds__(256) void k_convert_f32_f16(const float* __restrict__ in,
                                                         _Float16* __restrict__ out, long n) {
  long i = (long)blockIdx.x * 256 + threadIdx.x;
  if (i < n) out[i] = (_Float16)in[i];
}

__global__ __launch_bounds__(256) void k_zero_f32(float* __restrict__ p, long n) {
  long i = (long)blockIdx.x * 256 + threadIdx.x;
  if (i < n) p[i] = 0.0f;
}

// out[b][c][r] = in[b][r][c]   (f32 -> f16)
__global__ __launch_bounds__(256) void k_transpose_f32_f16(const float* __restrict__ in,
                                                           _Float16* __restrict__ out,
                                                           int batch, int rows, int cols) {
  long n = (long)batch * rows * cols;
  long i = (long)blockIdx.x * 256 + threadIdx.x;
  if (i >= n) return;
  int r = i % rows;
  long t = i / rows;
  int c = t % cols;
  int b = t / cols;
  out[i] = (_Float16)in[((long)b * rows + r) * cols + c];
}

// Mt[b][d][t] = M[t][b][d]   (f16 -> f16)
__global__ __launch_bounds__(256) void k_transpose_M(const _Float16* __restrict__ M,
                                                     _Float16* __restrict__ Mt) {
  long n = (long)BB * D2 * SS;
  long i = (long)blockIdx.x * 256 + threadIdx.x;
  if (i >= n) return;
  int t = i % SS;
  long q = i / SS;
  int d = q % D2;
  int b = q / D2;
  Mt[i] = M[((long)t * BB + b) * D2 + d];
}

// conv16[(t*B+b)*DH + d] = d<FIN ? features[n][d] : out2[n][d-FIN],  n = b*S + t
__global__ __launch_bounds__(256) void k_concat(const float* __restrict__ features,
                                                const _Float16* __restrict__ out2,
                                                _Float16* __restrict__ conv16) {
  long nAll = (long)NNODES * DH;
  long i = (long)blockIdx.x * 256 + threadIdx.x;
  if (i >= nAll) return;
  int d = i % DH;
  long row = i / DH;           // t*B + b
  int b = row % BB;
  int t = row / BB;
  long node = (long)b * SS + t;
  float v = (d < FIN) ? features[node * FIN + d] : (float)out2[node * HID + (d - FIN)];
  conv16[i] = (_Float16)v;
}

// ---------------------------------------------------------------------------
// Edge kernels
// ---------------------------------------------------------------------------
__global__ __launch_bounds__(256) void k_count_edges(const int* __restrict__ edge_index,
                                                     const int* __restrict__ edge_type,
                                                     float* __restrict__ cnt) {
  int e = blockIdx.x * 256 + threadIdx.x;
  if (e >= NEDGE) return;
  int dst = edge_index[NEDGE + e];
  int r = edge_type[e];
  atomicAdd(cnt + (long)dst * NREL + r, 1.0f);
}

// out[dst] += H[src][rel] / max(cnt[dst][rel],1)
__global__ __launch_bounds__(256) void k_rgcn_scatter(const _Float16* __restrict__ H,
                                                      const int* __restrict__ edge_index,
                                                      const int* __restrict__ edge_type,
                                                      const float* __restrict__ cnt,
                                                      float* __restrict__ outp) {
  int e = blockIdx.x;
  int d = threadIdx.x;
  int src = edge_index[e];
  int dst = edge_index[NEDGE + e];
  int r = edge_type[e];
  float v = (float)H[(long)src * (NREL * HID) + (long)r * HID + d];
  float c = cnt[(long)dst * NREL + r];
  atomicAdd(outp + (long)dst * HID + d, v / fmaxf(c, 1.0f));
}

// agg[dst] += out[src]
__global__ __launch_bounds__(256) void k_gc_scatter(const float* __restrict__ outp,
                                                    const int* __restrict__ edge_index,
                                                    float* __restrict__ agg) {
  int e = blockIdx.x;
  int d = threadIdx.x;
  int src = edge_index[e];
  int dst = edge_index[NEDGE + e];
  atomicAdd(agg + (long)dst * HID + d, outp[(long)src * HID + d]);
}

// ---------------------------------------------------------------------------
// Generic WMMA GEMM:  C[m,n] (+)= sum_k A[m,k] * Bt[n,k]   (f16 x f16 -> f32)
// Block tile 128x64, BK=64, 8 waves, each wave a 32x32 tile (2x2 WMMA frags).
// Global->LDS staging via TDM (tensor_load_to_lds, double buffered, overlapped
// with WMMA, synced with s_wait_tensorcnt) with hardware row padding; manual
// global/ds fallback if the builtin is unavailable.
// M multiple of 128, N multiple of 64, K multiple of 64.
// ---------------------------------------------------------------------------
__global__ __launch_bounds__(256) void gemm_wmma(
    const _Float16* __restrict__ A,  long aRow, long aBatch,
    const _Float16* __restrict__ Bt, long bRow, long bBatch,
    float* __restrict__ C32, _Float16* __restrict__ C16,
    const float* __restrict__ bias,
    long cRow, long cBatch, int K, int flags) {
  const _Float16* Ab = A  + (long)blockIdx.z * aBatch + (long)blockIdx.y * 128 * aRow;
  const _Float16* Bb = Bt + (long)blockIdx.z * bBatch + (long)blockIdx.x * 64 * bRow;

  __shared__ alignas(16) _Float16 As[2][128][72];   // 64 data halves + 8 pad
  __shared__ alignas(16) _Float16 Bs[2][64][72];

  const int tid = threadIdx.x;
  const int lane = tid & 31;
  const int wave = tid >> 5;
  const int wm = wave & 3;          // 4 m-subtiles of 32
  const int wn = wave >> 2;         // 2 n-subtiles of 32

  v8f acc[2][2] = {};

  const int cn  = lane & 15;
  const int kb  = (lane >> 4) * 8;   // A frag K sub-offset
  const int kb2 = (lane >> 4) * 16;  // unused; B frags also use kb
  const int ra0 = wm * 32 + cn;
  const int ra1 = ra0 + 16;
  const int nb0 = wn * 32 + cn;
  const int nb1 = nb0 + 16;

  const int nk = K >> 6;

  // ---- prologue: stage 0 ----
#if USE_TDM
  if (wave == 0) {
    tdm_load_2d(Ab, (unsigned)(uintptr_t)&As[0][0][0], aRow, 128);
    tdm_load_2d(Bb, (unsigned)(uintptr_t)&Bs[0][0][0], bRow, 64);
    __builtin_amdgcn_s_wait_tensorcnt(0);
  }
  __syncthreads();
#else
  for (int c = tid; c < 1024; c += 256) {
    int r = c >> 3, col = (c & 7) * 8;
    *(float4*)&As[0][r][col] = *(const float4*)(Ab + (long)r * aRow + col);
  }
  for (int c = tid; c < 512; c += 256) {
    int r = c >> 3, col = (c & 7) * 8;
    *(float4*)&Bs[0][r][col] = *(const float4*)(Bb + (long)r * bRow + col);
  }
  __syncthreads();
#endif

  int buf = 0;
  for (int s = 0; s < nk; ++s) {
    // ---- issue stage s+1 (overlaps with compute of stage s) ----
    if (s + 1 < nk) {
      const long k0n = (long)(s + 1) << 6;
#if USE_TDM
      if (wave == 0) {
        tdm_load_2d(Ab + k0n, (unsigned)(uintptr_t)&As[buf ^ 1][0][0], aRow, 128);
        tdm_load_2d(Bb + k0n, (unsigned)(uintptr_t)&Bs[buf ^ 1][0][0], bRow, 64);
      }
#else
      for (int c = tid; c < 1024; c += 256) {
        int r = c >> 3, col = (c & 7) * 8;
        *(float4*)&As[buf ^ 1][r][col] = *(const float4*)(Ab + (long)r * aRow + k0n + col);
      }
      for (int c = tid; c < 512; c += 256) {
        int r = c >> 3, col = (c & 7) * 8;
        *(float4*)&Bs[buf ^ 1][r][col] = *(const float4*)(Bb + (long)r * bRow + k0n + col);
      }
      if (s + 2 < nk) {
        __builtin_prefetch(Ab + ((long)(s + 2) << 6) + (long)(tid >> 3) * aRow, 0, 1);
        __builtin_prefetch(Bb + ((long)(s + 2) << 6) + (long)(tid >> 3) * bRow, 0, 1);
      }
#endif
    }
    // ---- compute stage s ----
#pragma unroll
    for (int kk = 0; kk < 64; kk += 32) {
      v16h a0, a1, b0, b1;
      ((float4*)&a0)[0] = *(const float4*)&As[buf][ra0][kk + kb];
      ((float4*)&a0)[1] = *(const float4*)&As[buf][ra0][kk + 16 + kb];
      ((float4*)&a1)[0] = *(const float4*)&As[buf][ra1][kk + kb];
      ((float4*)&a1)[1] = *(const float4*)&As[buf][ra1][kk + 16 + kb];
      ((float4*)&b0)[0] = *(const float4*)&Bs[buf][nb0][kk + kb];
      ((float4*)&b0)[1] = *(const float4*)&Bs[buf][nb0][kk + 16 + kb];
      ((float4*)&b1)[0] = *(const float4*)&Bs[buf][nb1][kk + kb];
      ((float4*)&b1)[1] = *(const float4*)&Bs[buf][nb1][kk + 16 + kb];
      acc[0][0] = WMMA16(a0, b0, acc[0][0]);
      acc[0][1] = WMMA16(a0, b1, acc[0][1]);
      acc[1][0] = WMMA16(a1, b0, acc[1][0]);
      acc[1][1] = WMMA16(a1, b1, acc[1][1]);
    }
#if USE_TDM
    if (wave == 0) __builtin_amdgcn_s_wait_tensorcnt(0);
#endif
    __syncthreads();
    buf ^= 1;
  }

  // ---- epilogue ----
  long rowG0 = (long)blockIdx.y * 128 + wm * 32;
  long colG0 = (long)blockIdx.x * 64 + wn * 32;
  int rAdd = (lane >> 4) * 8;
#pragma unroll
  for (int mi = 0; mi < 2; ++mi)
#pragma unroll
    for (int ni = 0; ni < 2; ++ni)
#pragma unroll
      for (int v = 0; v < 8; ++v) {
        float val = acc[mi][ni][v];
        long r = rowG0 + mi * 16 + v + rAdd;
        long c = colG0 + ni * 16 + cn;
        long idx = (long)blockIdx.z * cBatch + r * cRow + c;
        if (flags & FLAG_ACCUM)  val += C32[idx];
        if (flags & FLAG_BIAS)   val += bias[c];
        if (flags & FLAG_RELU)   val = fmaxf(val, 0.0f);
        if (flags & FLAG_OUTF32) C32[idx] = val;
        if (flags & FLAG_OUTF16) C16[idx] = (_Float16)val;
      }
}

// ---------------------------------------------------------------------------
// LSTM recurrent scan (one direction). Single 1024-thread workgroup (32 waves).
// P = x@Wih^T + b precomputed (S*B, G4) f32.  WhhT: (G4, DH) f16 (N x K).
// h kept in LDS (f16), c kept in WMMA C-fragment registers.
// Each wave owns 3 (m-tile, d-tile) pairs x 4 gates = 12 WMMA accumulators.
// ---------------------------------------------------------------------------
__global__ __launch_bounds__(1024) void lstm_scan(
    const float* __restrict__ P,
    const _Float16* __restrict__ WhhT,
    _Float16* __restrict__ hOut,     // (S*B, D2), already offset by dir*DH
    int rev) {
  __shared__ alignas(16) _Float16 hbuf[32][DH + 8];

  int tid = threadIdx.x;
  int lane = tid & 31;
  int wave = tid >> 5;

  for (int i = tid; i < 32 * (DH + 8); i += 1024)
    ((_Float16*)hbuf)[i] = (_Float16)0.0f;

  float cst[3][8];
#pragma unroll
  for (int p = 0; p < 3; ++p)
#pragma unroll
    for (int v = 0; v < 8; ++v) cst[p][v] = 0.0f;
  __syncthreads();

  const int cn = lane & 15;
  const int rAdd = (lane >> 4) * 8;
  const int kb = (lane >> 4) * 8;
  const int kb2 = (lane >> 4) * 16;

  for (int step = 0; step < SS; ++step) {
    int t = rev ? (SS - 1 - step) : step;
    const float* Pt = P + (long)t * BB * G4;

    v8f acc[3][4];
    // init accumulators from precomputed input projection
#pragma unroll
    for (int p = 0; p < 3; ++p) {
      int pid = wave + 32 * p;      // 0..95
      int m = pid & 1;
      int dt = pid >> 1;            // 0..47
#pragma unroll
      for (int g = 0; g < 4; ++g)
#pragma unroll
        for (int v = 0; v < 8; ++v) {
          int b = m * 16 + v + rAdd;
          acc[p][g][v] = Pt[(long)b * G4 + g * DH + dt * 16 + cn];
        }
    }

    // recurrent GEMM: acc += h @ Whh^T
#pragma unroll
    for (int p = 0; p < 3; ++p) {
      int pid = wave + 32 * p;
      int m = pid & 1;
      int dt = pid >> 1;
      int arow = m * 16 + cn;
      for (int k = 0; k < DH; k += 32) {
        v16h a;
        ((float4*)&a)[0] = *(const float4*)&hbuf[arow][k + kb];
        ((float4*)&a)[1] = *(const float4*)&hbuf[arow][k + 16 + kb];
#pragma unroll
        for (int g = 0; g < 4; ++g) {
          int nrow = g * DH + dt * 16 + cn;
          v16h b;
          const _Float16* bp = WhhT + (long)nrow * DH + k + kb;
          ((float4*)&b)[0] = *(const float4*)bp;
          ((float4*)&b)[1] = *(const float4*)(bp + 16);
          acc[p][g] = WMMA16(a, b, acc[p][g]);
        }
      }
    }
    __syncthreads();   // all waves done reading hbuf

    // gates + state update directly on C fragments; write new h
#pragma unroll
    for (int p = 0; p < 3; ++p) {
      int pid = wave + 32 * p;
      int m = pid & 1;
      int dt = pid >> 1;
#pragma unroll
      for (int v = 0; v < 8; ++v) {
        float ig = sigmoidf_(acc[p][0][v]);
        float fg = sigmoidf_(acc[p][1][v]);
        float gg = tanhf(acc[p][2][v]);
        float og = sigmoidf_(acc[p][3][v]);
        float ct = fg * cst[p][v] + ig * gg;
        cst[p][v] = ct;
        float ht = og * tanhf(ct);
        int b = m * 16 + v + rAdd;
        int d = dt * 16 + cn;
        _Float16 hh = (_Float16)ht;
        hbuf[b][d] = hh;
        hOut[((long)t * BB + b) * D2 + d] = hh;
      }
    }
    __syncthreads();   // writes visible before next step's reads
  }
}

// ---------------------------------------------------------------------------
// Masked tanh-softmax over s: alpha[b][t][s]
// ---------------------------------------------------------------------------
__global__ __launch_bounds__(128) void k_softmax(const float* __restrict__ scores,
                                                 const float* __restrict__ umask,
                                                 _Float16* __restrict__ alpha) {
  int bx = blockIdx.x;        // b*S + t
  int b = bx >> 7;
  int s = threadIdx.x;
  __shared__ float red[128];

  float m = umask[(long)b * SS + s];
  float v = tanhf(scores[(long)bx * SS + s] * m);

  red[s] = v;
  __syncthreads();
  for (int o = 64; o > 0; o >>= 1) {
    if (s < o) red[s] = fmaxf(red[s], red[s + o]);
    __syncthreads();
  }
  float mx = red[0];
  __syncthreads();

  float e = __expf(v - mx) * m;
  red[s] = e;
  __syncthreads();
  for (int o = 64; o > 0; o >>= 1) {
    if (s < o) red[s] += red[s + o];
    __syncthreads();
  }
  float sum = red[0];
  alpha[(long)bx * SS + s] = (_Float16)(e / fmaxf(sum, 1e-20f));
}

// ---------------------------------------------------------------------------
// Host-side launch helper
// ---------------------------------------------------------------------------
static inline void launch_gemm(hipStream_t st,
                               const _Float16* A, long aR, long aB,
                               const _Float16* Bt, long bR, long bB,
                               float* C32, _Float16* C16, const float* bias,
                               long cR, long cB, int M, int N, int K, int batch, int flags) {
  dim3 grid(N / 64, M / 128, batch);
  hipLaunchKernelGGL(gemm_wmma, grid, dim3(256), 0, st,
                     A, aR, aB, Bt, bR, bB, C32, C16, bias, cR, cB, K, flags);
}

static inline dim3 grid1d(long n) { return dim3((unsigned)((n + 255) / 256)); }

extern "C" void kernel_launch(void* const* d_in, const int* in_sizes, int n_in,
                              void* d_out, int out_size, void* d_ws, size_t ws_size,
                              hipStream_t stream) {
  const float* features   = (const float*)d_in[0];
  const int*   edge_index = (const int*)d_in[1];
  const int*   edge_type  = (const int*)d_in[2];
  const float* umask      = (const float*)d_in[4];
  const float* rgcn_Wrel  = (const float*)d_in[5];
  const float* rgcn_Wroot = (const float*)d_in[6];
  const float* rgcn_b     = (const float*)d_in[7];
  const float* gc_Wrel    = (const float*)d_in[8];
  const float* gc_Wroot   = (const float*)d_in[9];
  const float* gc_b       = (const float*)d_in[10];
  const float* Wih0       = (const float*)d_in[11];
  const float* Whh0       = (const float*)d_in[12];
  const float* b0         = (const float*)d_in[13];
  const float* Wih1       = (const float*)d_in[14];
  const float* Whh1       = (const float*)d_in[15];
  const float* b1         = (const float*)d_in[16];
  const float* matt_W     = (const float*)d_in[17];
  const float* matt_b     = (const float*)d_in[18];
  const float* lin_W      = (const float*)d_in[19];
  const float* lin_b      = (const float*)d_in[20];
  float* out              = (float*)d_out;

  // bump allocator over workspace
  size_t off = 0;
  char* wsb = (char*)d_ws;
  auto alloc = [&](size_t bytes) -> void* {
    void* p = wsb + off;
    off += (bytes + 255) & ~(size_t)255;
    return p;
  };

  _Float16* feat16    = (_Float16*)alloc((size_t)NNODES * FIN * 2);
  _Float16* wrelT16   = (_Float16*)alloc((size_t)NREL * HID * FIN * 2);
  _Float16* wrootT16  = (_Float16*)alloc((size_t)HID * FIN * 2);
  _Float16* H16       = (_Float16*)alloc((size_t)NNODES * NREL * HID * 2);
  float*    cnt       = (float*)   alloc((size_t)NNODES * NREL * 4);
  float*    rgout     = (float*)   alloc((size_t)NNODES * HID * 4);
  _Float16* rgout16   = (_Float16*)alloc((size_t)NNODES * HID * 2);
  float*    agg       = (float*)   alloc((size_t)NNODES * HID * 4);
  _Float16* agg16     = (_Float16*)alloc((size_t)NNODES * HID * 2);
  _Float16* gcWrelT16 = (_Float16*)alloc((size_t)HID * HID * 2);
  _Float16* gcWrootT16= (_Float16*)alloc((size_t)HID * HID * 2);
  float*    gtmp      = (float*)   alloc((size_t)NNODES * HID * 4);
  _Float16* out2_16   = (_Float16*)alloc((size_t)NNODES * HID * 2);
  _Float16* conv16    = (_Float16*)alloc((size_t)NNODES * DH * 2);
  _Float16* wih0_16   = (_Float16*)alloc((size_t)2 * G4 * DH * 2);
  _Float16* whh0_16   = (_Float16*)alloc((size_t)2 * G4 * DH * 2);
  _Float16* wih1_16   = (_Float16*)alloc((size_t)2 * G4 * D2 * 2);
  _Float16* whh1_16   = (_Float16*)alloc((size_t)2 * G4 * DH * 2);
  float*    Pbuf      = (float*)   alloc((size_t)NNODES * G4 * 4);
  _Float16* h1_16     = (_Float16*)alloc((size_t)NNODES * D2 * 2);
  _Float16* M16       = (_Float16*)alloc((size_t)NNODES * D2 * 2);
  _Float16* mattW16   = (_Float16*)alloc((size_t)D2 * D2 * 2);
  _Float16* X16       = (_Float16*)alloc((size_t)NNODES * D2 * 2);
  float*    scoresB   = (float*)   alloc((size_t)BB * SS * SS * 4);
  _Float16* alpha16   = (_Float16*)alloc((size_t)BB * SS * SS * 2);
  _Float16* Mt16      = (_Float16*)alloc((size_t)BB * D2 * SS * 2);
  _Float16* att16     = (_Float16*)alloc((size_t)NNODES * D2 * 2);
  _Float16* linW16    = (_Float16*)alloc((size_t)DH * D2 * 2);
  (void)ws_size; (void)in_sizes; (void)n_in; (void)out_size;

  // ---- weight / activation conversions -------------------------------------
  hipLaunchKernelGGL(k_convert_f32_f16, grid1d((long)NNODES * FIN), dim3(256), 0, stream,
                     features, feat16, (long)NNODES * FIN);
  hipLaunchKernelGGL(k_transpose_f32_f16, grid1d((long)NREL * FIN * HID), dim3(256), 0, stream,
                     rgcn_Wrel, wrelT16, NREL, FIN, HID);
  hipLaunchKernelGGL(k_transpose_f32_f16, grid1d((long)FIN * HID), dim3(256), 0, stream,
                     rgcn_Wroot, wrootT16, 1, FIN, HID);
  hipLaunchKernelGGL(k_transpose_f32_f16, grid1d((long)HID * HID), dim3(256), 0, stream,
                     gc_Wrel, gcWrelT16, 1, HID, HID);
  hipLaunchKernelGGL(k_transpose_f32_f16, grid1d((long)HID * HID), dim3(256), 0, stream,
                     gc_Wroot, gcWrootT16, 1, HID, HID);
  hipLaunchKernelGGL(k_convert_f32_f16, grid1d((long)2 * G4 * DH), dim3(256), 0, stream,
                     Wih0, wih0_16, (long)2 * G4 * DH);
  hipLaunchKernelGGL(k_convert_f32_f16, grid1d((long)2 * G4 * DH), dim3(256), 0, stream,
                     Whh0, whh0_16, (long)2 * G4 * DH);
  hipLaunchKernelGGL(k_convert_f32_f16, grid1d((long)2 * G4 * D2), dim3(256), 0, stream,
                     Wih1, wih1_16, (long)2 * G4 * D2);
  hipLaunchKernelGGL(k_convert_f32_f16, grid1d((long)2 * G4 * DH), dim3(256), 0, stream,
                     Whh1, whh1_16, (long)2 * G4 * DH);
  hipLaunchKernelGGL(k_convert_f32_f16, grid1d((long)D2 * D2), dim3(256), 0, stream,
                     matt_W, mattW16, (long)D2 * D2);
  hipLaunchKernelGGL(k_convert_f32_f16, grid1d((long)DH * D2), dim3(256), 0, stream,
                     lin_W, linW16, (long)DH * D2);

  // ---- RGCN ----------------------------------------------------------------
  // root: rgout = features @ Wroot + b
  launch_gemm(stream, feat16, FIN, 0, wrootT16, FIN, 0, rgout, nullptr, rgcn_b,
              HID, 0, NNODES, HID, FIN, 1, FLAG_BIAS | FLAG_OUTF32);
  // all relations fused: H16 = features @ [Wrel_0 .. Wrel_15]
  launch_gemm(stream, feat16, FIN, 0, wrelT16, FIN, 0, nullptr, H16, nullptr,
              NREL * HID, 0, NNODES, NREL * HID, FIN, 1, FLAG_OUTF16);
  hipLaunchKernelGGL(k_zero_f32, grid1d((long)NNODES * NREL), dim3(256), 0, stream,
                     cnt, (long)NNODES * NREL);
  hipLaunchKernelGGL(k_count_edges, dim3(NEDGE / 256), dim3(256), 0, stream,
                     edge_index, edge_type, cnt);
  hipLaunchKernelGGL(k_rgcn_scatter, dim3(NEDGE), dim3(HID), 0, stream,
                     H16, edge_index, edge_type, cnt, rgout);

  // ---- GraphConv -----------------------------------------------------------
  hipLaunchKernelGGL(k_convert_f32_f16, grid1d((long)NNODES * HID), dim3(256), 0, stream,
                     rgout, rgout16, (long)NNODES * HID);
  hipLaunchKernelGGL(k_zero_f32, grid1d((long)NNODES * HID), dim3(256), 0, stream,
                     agg, (long)NNODES * HID);
  hipLaunchKernelGGL(k_gc_scatter, dim3(NEDGE), dim3(HID), 0, stream,
                     rgout, edge_index, agg);
  hipLaunchKernelGGL(k_convert_f32_f16, grid1d((long)NNODES * HID), dim3(256), 0, stream,
                     agg, agg16, (long)NNODES * HID);
  launch_gemm(stream, agg16, HID, 0, gcWrelT16, HID, 0, gtmp, nullptr, gc_b,
              HID, 0, NNODES, HID, HID, 1, FLAG_BIAS | FLAG_OUTF32);
  launch_gemm(stream, rgout16, HID, 0, gcWrootT16, HID, 0, gtmp, out2_16, nullptr,
              HID, 0, NNODES, HID, HID, 1, FLAG_ACCUM | FLAG_OUTF16);

  // ---- concat + reshape ----------------------------------------------------
  hipLaunchKernelGGL(k_concat, grid1d((long)NNODES * DH), dim3(256), 0, stream,
                     features, out2_16, conv16);

  // ---- BiLSTM layer 0 ------------------------------------------------------
  for (int dir = 0; dir < 2; ++dir) {
    launch_gemm(stream, conv16, DH, 0, wih0_16 + (size_t)dir * G4 * DH, DH, 0,
                Pbuf, nullptr, b0 + (size_t)dir * G4, G4, 0,
                NNODES, G4, DH, 1, FLAG_BIAS | FLAG_OUTF32);
    hipLaunchKernelGGL(lstm_scan, dim3(1), dim3(1024), 0, stream,
                       Pbuf, whh0_16 + (size_t)dir * G4 * DH, h1_16 + (size_t)dir * DH, dir);
  }
  // ---- BiLSTM layer 1 ------------------------------------------------------
  for (int dir = 0; dir < 2; ++dir) {
    launch_gemm(stream, h1_16, D2, 0, wih1_16 + (size_t)dir * G4 * D2, D2, 0,
                Pbuf, nullptr, b1 + (size_t)dir * G4, G4, 0,
                NNODES, G4, D2, 1, FLAG_BIAS | FLAG_OUTF32);
    hipLaunchKernelGGL(lstm_scan, dim3(1), dim3(1024), 0, stream,
                       Pbuf, whh1_16 + (size_t)dir * G4 * DH, M16 + (size_t)dir * DH, dir);
  }

  // ---- Matching attention --------------------------------------------------
  // X = M @ matt_W^T + matt_b
  launch_gemm(stream, M16, D2, 0, mattW16, D2, 0, nullptr, X16, matt_b,
              D2, 0, NNODES, D2, D2, 1, FLAG_BIAS | FLAG_OUTF16);
  // scores[b][t][s] = X[t,b,:] . M[s,b,:]   (batched over b)
  launch_gemm(stream, X16, (long)BB * D2, D2, M16, (long)BB * D2, D2,
              scoresB, nullptr, nullptr, SS, (long)SS * SS,
              SS, SS, D2, BB, FLAG_OUTF32);
  hipLaunchKernelGGL(k_softmax, dim3(BB * SS), dim3(SS), 0, stream,
                     scoresB, umask, alpha16);
  hipLaunchKernelGGL(k_transpose_M, grid1d((long)BB * D2 * SS), dim3(256), 0, stream,
                     M16, Mt16);
  // att[t,b,:] = sum_s alpha[b][t][s] * M[s,b,:]
  launch_gemm(stream, alpha16, SS, (long)SS * SS, Mt16, SS, (long)D2 * SS,
              nullptr, att16, nullptr, (long)BB * D2, D2,
              SS, D2, SS, BB, FLAG_OUTF16);
  // hidden = relu(att @ lin_W^T + lin_b)  -> d_out (S,B,DH) f32
  launch_gemm(stream, att16, D2, 0, linW16, D2, 0, out, nullptr, lin_b,
              DH, 0, NNODES, DH, D2, 1, FLAG_BIAS | FLAG_RELU | FLAG_OUTF32);
}
